// DeformableGCN_9844065042903
// MI455X (gfx1250) — matrix operand order repaired and checked by reading, not compiled, not run on hardware
//
#include <hip/hip_runtime.h>
#include <hip/hip_bf16.h>

typedef __attribute__((ext_vector_type(2)))  float    v2f;
typedef __attribute__((ext_vector_type(8)))  float    v8f;
typedef __attribute__((ext_vector_type(16))) _Float16 v16h;

// ---------------------------------------------------------------------------
// Degree: atomicAdd 1.0 per edge into deg[dst]
// ---------------------------------------------------------------------------
__global__ void dgcn_edge_deg(const int* __restrict__ dst, float* __restrict__ deg, int E) {
    int e = blockIdx.x * blockDim.x + threadIdx.x;
    if (e >= E) return;
    atomicAdd(&deg[dst[e]], 1.0f);
}

__global__ void dgcn_recip_clamp(float* __restrict__ deg, int n) {
    int i = blockIdx.x * blockDim.x + threadIdx.x;
    if (i >= n) return;
    deg[i] = 1.0f / fmaxf(deg[i], 1.0f);
}

// ---------------------------------------------------------------------------
// Edge scatter: out[dst] += w * Z[src]  (w = scores[e], or 1.0 if scores==null)
// One thread per (edge, 4-float chunk); chunks = D/4 = 1<<shift.
// float4 gather + 4 scalar f32 atomics (global_atomic_add_f32).
// ---------------------------------------------------------------------------
__global__ void dgcn_edge_scatter(const int* __restrict__ src, const int* __restrict__ dst,
                                  const float* __restrict__ scores,
                                  const float* __restrict__ Z, float* __restrict__ out,
                                  int E, int D, int shift) {
    long tid = (long)blockIdx.x * blockDim.x + threadIdx.x;
    int e = (int)(tid >> shift);
    if (e >= E) return;
    int c = (int)(tid & ((1 << shift) - 1));
    int s = src[e];
    int d = dst[e];
    float w = scores ? scores[e] : 1.0f;
    const float4 z = *reinterpret_cast<const float4*>(Z + (size_t)s * D + c * 4);
    float* o = out + (size_t)d * D + c * 4;
    atomicAdd(o + 0, w * z.x);
    atomicAdd(o + 1, w * z.y);
    atomicAdd(o + 2, w * z.z);
    atomicAdd(o + 3, w * z.w);
}

// ---------------------------------------------------------------------------
// Smoothing pointwise: v = hn[i]*inv[node]; h[i]=v; acc[i]+=v   (d==64)
// ---------------------------------------------------------------------------
__global__ void dgcn_smooth_update(const float* __restrict__ hn, const float* __restrict__ inv,
                                   float* __restrict__ h, float* __restrict__ acc, int total) {
    int i = blockIdx.x * blockDim.x + threadIdx.x;
    if (i >= total) return;
    float v = hn[i] * inv[i >> 6];
    h[i]   = v;
    acc[i] += v;
}

__global__ void dgcn_scale(float* __restrict__ p, float s, int n) {
    int i = blockIdx.x * blockDim.x + threadIdx.x;
    if (i < n) p[i] *= s;
}

__global__ void dgcn_relu(float* __restrict__ p, int n) {
    int i = blockIdx.x * blockDim.x + threadIdx.x;
    if (i < n) p[i] = fmaxf(p[i], 0.0f);
}

// ---------------------------------------------------------------------------
// Per-node attention projections: asrc[i] = H[i,:]·Watt[0:d], adst[i] = H[i,:]·Watt[d:2d]
// One wave32 per node, shuffle reduction.
// ---------------------------------------------------------------------------
__global__ void dgcn_att_parts(const float* __restrict__ H, const float* __restrict__ Watt,
                               float* __restrict__ asrc, float* __restrict__ adst,
                               int n, int d) {
    int node = blockIdx.x * blockDim.y + threadIdx.y;
    if (node >= n) return;
    int lane = threadIdx.x;
    float s = 0.0f, t = 0.0f;
    for (int f = lane; f < d; f += 32) {
        float v = H[(size_t)node * d + f];
        s += v * Watt[f];
        t += v * Watt[d + f];
    }
    for (int off = 16; off > 0; off >>= 1) {
        s += __shfl_down(s, off, 32);
        t += __shfl_down(t, off, 32);
    }
    if (lane == 0) { asrc[node] = s; adst[node] = t; }
}

// score[e] = leaky_relu(asrc[src]+adst[dst]+b, 0.01)
__global__ void dgcn_edge_scores(const int* __restrict__ src, const int* __restrict__ dst,
                                 const float* __restrict__ asrc, const float* __restrict__ adst,
                                 const float* __restrict__ bias, float* __restrict__ scores,
                                 int E) {
    int e = blockIdx.x * blockDim.x + threadIdx.x;
    if (e >= E) return;
    float v = asrc[src[e]] + adst[dst[e]] + bias[0];
    scores[e] = (v > 0.0f) ? v : 0.01f * v;
}

// ---------------------------------------------------------------------------
// WMMA GEMM: C[M,Nc] = A[M,K] @ B[K,Nc], all row-major fp32.
// One wave32 per 16x16 C tile; grid = (M/16 tiles, Nc/16 tiles), block = 32.
// fp32 path: V_WMMA_F32_16X16X4_F32, 16 MACs over K=64.
// Lane layouts per CDNA5 ISA §7.12.2:
//   A 16x4 f32 : lane(l&15)=M, v0 holds K=(k0+2*(l>>4)), v1 holds K+1.
//   C/D 16x16  : VGPR r -> row r+8*(l>>4), col = l&15.
//   B 4x16     : assumed dual of A (v0: K=k0+2*hi, v1: K+1, col = l&15).
// ---------------------------------------------------------------------------
__global__ void dgcn_wmma_gemm(const float* __restrict__ A, const float* __restrict__ B,
                               float* __restrict__ C, int M, int K, int Nc) {
    int rt   = blockIdx.x;
    int ct   = blockIdx.y;
    int lane = threadIdx.x;
    int m    = lane & 15;
    int hi   = lane >> 4;

    int row = rt * 16 + m;
    int arow = row < M ? row : (M - 1);   // clamp (M is a multiple of 16 here anyway)
    int col = ct * 16 + m;

    v8f acc = {};

#if __has_builtin(__builtin_amdgcn_wmma_f32_16x16x4_f32)
    for (int k0 = 0; k0 < K; k0 += 4) {
        int ka = k0 + 2 * hi;
        v2f a, b;
        a.x = A[(size_t)arow * K + ka];
        a.y = A[(size_t)arow * K + ka + 1];
        b.x = B[(size_t)(ka)     * Nc + col];
        b.y = B[(size_t)(ka + 1) * Nc + col];
        acc = __builtin_amdgcn_wmma_f32_16x16x4_f32(
            /*neg_a=*/false, a, /*neg_b=*/false, b,
            /*c_mod=*/(short)0, acc, /*reuse_a=*/false, /*reuse_b=*/false);
    }
#else
    // Fallback: f16 inputs, fp32 accumulate (V_WMMA_F32_16X16X32_F16).
    for (int k0 = 0; k0 < K; k0 += 32) {
        v16h a, b;
        #pragma unroll
        for (int v = 0; v < 8; ++v) {
            #pragma unroll
            for (int e = 0; e < 2; ++e) {
                int ka = k0 + ((v < 4) ? 0 : 16) + 2 * (v & 3) + e + hi * 8;
                int kb = k0 + ((v < 4) ? 0 : 16) + 2 * (v & 3) + e + hi * 8;
                a[2 * v + e] = (_Float16)A[(size_t)arow * K + ka];
                b[2 * v + e] = (_Float16)B[(size_t)kb * Nc + col];
            }
        }
        acc = __builtin_amdgcn_wmma_f32_16x16x32_f16(
            false, a, false, b, (short)0, acc, false, false);
    }
#endif

    #pragma unroll
    for (int r = 0; r < 8; ++r) {
        int orow = rt * 16 + r + 8 * hi;
        if (orow < M) C[(size_t)orow * Nc + ct * 16 + m] = acc[r];
    }
}

// ---------------------------------------------------------------------------
// Orchestration
// ---------------------------------------------------------------------------
extern "C" void kernel_launch(void* const* d_in, const int* in_sizes, int n_in,
                              void* d_out, int out_size, void* d_ws, size_t ws_size,
                              hipStream_t stream) {
    (void)n_in; (void)out_size; (void)ws_size;

    const float* x      = (const float*)d_in[0];   // [N,64]
    const int*   ei     = (const int*)  d_in[1];   // [2,E]
    const float* Watt1  = (const float*)d_in[2];   // [128,1]
    const float* batt1  = (const float*)d_in[3];   // [1]
    const float* Wlin1  = (const float*)d_in[4];   // [64,64]
    const float* Watt2  = (const float*)d_in[5];   // [128,1]
    const float* batt2  = (const float*)d_in[6];   // [1]
    const float* Wlin2  = (const float*)d_in[7];   // [64,32]

    const int N = in_sizes[0] / 64;
    const int E = in_sizes[1] / 2;
    const int DIN = 64, DOUT = 32;

    const int* srcp = ei;
    const int* dstp = ei + E;

    // Workspace layout (floats): 3 * N*64 feature buffers + deg + att parts + scores
    float* wf    = (float*)d_ws;
    size_t nd    = (size_t)N * DIN;
    float* h     = wf;              // smoothing state / later Z1 / later Z2
    float* hn    = h   + nd;        // scatter target / later feats1 -> h1
    float* acc   = hn  + nd;        // accumulator -> xs
    float* inv   = acc + nd;        // deg, then 1/max(deg,1)
    float* asrc  = inv + N;
    float* adst  = asrc + N;
    float* scores = adst + N;       // E floats

    const int TB = 256;
    dim3 blk(TB);

    // --- degree + reciprocal -------------------------------------------------
    hipMemsetAsync(inv, 0, (size_t)N * sizeof(float), stream);
    dgcn_edge_deg<<<dim3((E + TB - 1) / TB), blk, 0, stream>>>(dstp, inv, E);
    dgcn_recip_clamp<<<dim3((N + TB - 1) / TB), blk, 0, stream>>>(inv, N);

    // --- graph smoothing: h = x; acc = x; 3x {hn = scatter(h); h = hn*inv; acc += h}
    hipMemcpyAsync(h,   x, nd * sizeof(float), hipMemcpyDeviceToDevice, stream);
    hipMemcpyAsync(acc, x, nd * sizeof(float), hipMemcpyDeviceToDevice, stream);

    long tot64 = (long)E << 4;   // E * 16 chunk-threads (D=64)
    for (int step = 0; step < 3; ++step) {
        hipMemsetAsync(hn, 0, nd * sizeof(float), stream);
        dgcn_edge_scatter<<<dim3((unsigned)((tot64 + TB - 1) / TB)), blk, 0, stream>>>(
            srcp, dstp, nullptr, h, hn, E, DIN, 4);
        dgcn_smooth_update<<<dim3(((int)nd + TB - 1) / TB), blk, 0, stream>>>(
            hn, inv, h, acc, (int)nd);
    }
    // xs = acc / 4  (in place)
    dgcn_scale<<<dim3(((int)nd + TB - 1) / TB), blk, 0, stream>>>(acc, 0.25f, (int)nd);

    // --- layer 1: Z1 = xs @ W_lin1 (WMMA), att parts, scores, weighted scatter
    dgcn_wmma_gemm<<<dim3((N + 15) / 16, DIN / 16), dim3(32), 0, stream>>>(
        acc, Wlin1, h /*Z1*/, N, DIN, DIN);
    dgcn_att_parts<<<dim3((N + 7) / 8), dim3(32, 8), 0, stream>>>(
        acc, Watt1, asrc, adst, N, DIN);
    dgcn_edge_scores<<<dim3((E + TB - 1) / TB), blk, 0, stream>>>(
        srcp, dstp, asrc, adst, batt1, scores, E);
    hipMemsetAsync(hn, 0, nd * sizeof(float), stream);
    dgcn_edge_scatter<<<dim3((unsigned)((tot64 + TB - 1) / TB)), blk, 0, stream>>>(
        srcp, dstp, scores, h /*Z1*/, hn /*feats1*/, E, DIN, 4);
    dgcn_relu<<<dim3(((int)nd + TB - 1) / TB), blk, 0, stream>>>(hn, (int)nd);   // h1

    // --- layer 2: Z2 = h1 @ W_lin2 (WMMA), att parts, scores, scatter into d_out
    dgcn_wmma_gemm<<<dim3((N + 15) / 16, DOUT / 16), dim3(32), 0, stream>>>(
        hn, Wlin2, h /*Z2: N x 32*/, N, DIN, DOUT);
    dgcn_att_parts<<<dim3((N + 7) / 8), dim3(32, 8), 0, stream>>>(
        hn, Watt2, asrc, adst, N, DIN);
    dgcn_edge_scores<<<dim3((E + TB - 1) / TB), blk, 0, stream>>>(
        srcp, dstp, asrc, adst, batt2, scores, E);

    hipMemsetAsync(d_out, 0, (size_t)N * DOUT * sizeof(float), stream);
    long tot32 = (long)E << 3;   // E * 8 chunk-threads (D=32)
    dgcn_edge_scatter<<<dim3((unsigned)((tot32 + TB - 1) / TB)), blk, 0, stream>>>(
        srcp, dstp, scores, h /*Z2*/, (float*)d_out, E, DOUT, 3);
}
